// Som_79534204388018
// MI455X (gfx1250) — matrix operand order repaired and checked by reading, not compiled, not run on hardware
//
#include <hip/hip_runtime.h>
#include <math.h>

typedef _Float16 v16h __attribute__((ext_vector_type(16)));
typedef _Float16 v8h  __attribute__((ext_vector_type(8)));
typedef float    v8f  __attribute__((ext_vector_type(8)));
typedef int      v4i  __attribute__((ext_vector_type(4)));

#define BATCH 2048
#define FEAT  256
#define NW    16384   // rows*cols = 128*128
#define COLSN 128
#define LDT   272     // LDS K-stride in halfs (256 + 16 pad; keeps 32B align)

// ---- CDNA5 async global->LDS staging (ASYNCcnt path), with fallback --------
#if defined(__has_builtin)
# if __has_builtin(__builtin_amdgcn_global_load_async_to_lds_b128)
#  define USE_ASYNC_LDS 1
# endif
# if __has_builtin(__builtin_amdgcn_s_wait_asynccnt)
#  define HAVE_WAIT_ASYNC_BUILTIN 1
# endif
#endif

__device__ __forceinline__ void stage16(const _Float16* __restrict__ g,
                                        _Float16* l) {
#if defined(USE_ASYNC_LDS)
  typedef __attribute__((address_space(1))) v4i gv4i;
  typedef __attribute__((address_space(3))) v4i lv4i;
  __builtin_amdgcn_global_load_async_to_lds_b128((gv4i*)g, (lv4i*)l, 0, 0);
#else
  *(uint4*)l = *(const uint4*)g;
#endif
}

__device__ __forceinline__ void wait_async_done() {
#if defined(USE_ASYNC_LDS)
# if defined(HAVE_WAIT_ASYNC_BUILTIN)
  __builtin_amdgcn_s_wait_asynccnt(0);
# else
  asm volatile("s_wait_asynccnt 0x0" ::: "memory");
# endif
#endif
}

// ---------------------------------------------------------------------------
// Kernel 0: f32 -> f16 convert (into workspace) + per-row squared norms.
// One block per row (x rows first, then weight rows), 256 threads = 1 elem ea.
// ---------------------------------------------------------------------------
__global__ __launch_bounds__(256) void convert_norms(
    const float* __restrict__ x, const float* __restrict__ w,
    _Float16* __restrict__ xh, _Float16* __restrict__ wh,
    float* __restrict__ xn, float* __restrict__ wn)
{
  __shared__ float red[256];
  const int r = blockIdx.x;
  const int t = threadIdx.x;
  const float* src;
  _Float16* dsth;
  float* dstn;
  if (r < BATCH) {
    src = x + (size_t)r * FEAT; dsth = xh + (size_t)r * FEAT; dstn = xn + r;
  } else {
    const int q = r - BATCH;
    src = w + (size_t)q * FEAT; dsth = wh + (size_t)q * FEAT; dstn = wn + q;
  }
  const float v = src[t];
  dsth[t] = (_Float16)v;
  red[t] = v * v;
  __syncthreads();
  #pragma unroll
  for (int o = 128; o > 0; o >>= 1) {
    if (t < o) red[t] += red[t + o];
    __syncthreads();
  }
  if (t == 0) *dstn = red[0];
}

// ---------------------------------------------------------------------------
// Kernel 1: fused GEMM + distance epilogue.
// Block tile: 64(M=batch) x 128(N=neurons); 256 threads = 8 wave32s.
// Wave tile: 32x32 as 2x2 WMMA 16x16 (f16 inputs, f32 accumulate).
// Full K=256 staged to (dynamic) LDS once via async global->LDS copies,
// then 8 uninterrupted WMMA k-steps. LDS = 192*272*2 = 104,448 B.
// ---------------------------------------------------------------------------
__global__ __launch_bounds__(256) void som_dist(
    const _Float16* __restrict__ xh, const _Float16* __restrict__ wh,
    const float* __restrict__ xn, const float* __restrict__ wn,
    float* __restrict__ dist)
{
  extern __shared__ _Float16 smem[];
  _Float16* As = smem;             // 64  x LDT
  _Float16* Ws = smem + 64 * LDT;  // 128 x LDT

  const int tid  = threadIdx.x;
  const int lane = tid & 31;
  const int wid  = tid >> 5;
  const int m0 = blockIdx.x * 64;      // batch-tile base
  const int n0 = blockIdx.y * 128;     // neuron-tile base

  const int mw = (wid & 1) * 32;       // wave offset in M (2 waves)
  const int nw = (wid >> 1) * 32;      // wave offset in N (4 waves)
  const int halfsel = lane >> 4;       // 0: lanes 0-15, 1: lanes 16-31
  const int lrow = lane & 15;

  // Stage A: 64 rows x 256 halfs = 2048 x b128, 8 per thread (async -> LDS)
  #pragma unroll
  for (int i = 0; i < 8; ++i) {
    const int c = tid + 256 * i;
    const int row = c >> 5;
    const int off = (c & 31) * 8;
    stage16(&xh[(size_t)(m0 + row) * FEAT + off], &As[row * LDT + off]);
  }
  // Stage W: 128 rows x 256 halfs = 4096 x b128, 16 per thread
  #pragma unroll
  for (int i = 0; i < 16; ++i) {
    const int c = tid + 256 * i;
    const int row = c >> 5;
    const int off = (c & 31) * 8;
    stage16(&wh[(size_t)(n0 + row) * FEAT + off], &Ws[row * LDT + off]);
  }
  wait_async_done();
  __syncthreads();

  v8f acc00 = {}, acc01 = {}, acc10 = {}, acc11 = {};

  // 16-bit fragment K-offsets per half-wave (contiguous k = 16h + s, same for A and B):
  const int kA = halfsel * 16;         // A 16x32: lanes16-31 hold K=16..31
  const int kB = halfsel * 16;         // B 32x16: lanes16-31 hold K=16..31

  #pragma unroll
  for (int k0 = 0; k0 < FEAT; k0 += 32) {
    const _Float16* a0p = &As[(mw +      lrow) * LDT + k0 + kA];
    const _Float16* a1p = &As[(mw + 16 + lrow) * LDT + k0 + kA];
    const _Float16* b0p = &Ws[(nw +      lrow) * LDT + k0 + kB];
    const _Float16* b1p = &Ws[(nw + 16 + lrow) * LDT + k0 + kB];

    const v8h a0lo = *(const v8h*)(a0p);      // K = kA .. kA+7
    const v8h a0hi = *(const v8h*)(a0p + 8);  // K = kA+8 .. kA+15
    const v8h a1lo = *(const v8h*)(a1p);
    const v8h a1hi = *(const v8h*)(a1p + 8);
    const v8h b0lo = *(const v8h*)(b0p);      // K = kB .. kB+7
    const v8h b0hi = *(const v8h*)(b0p + 8);  // K = kB+8 .. kB+15
    const v8h b1lo = *(const v8h*)(b1p);
    const v8h b1hi = *(const v8h*)(b1p + 8);

    const v16h a0 = __builtin_shufflevector(a0lo, a0hi,
        0,1,2,3,4,5,6,7,8,9,10,11,12,13,14,15);
    const v16h a1 = __builtin_shufflevector(a1lo, a1hi,
        0,1,2,3,4,5,6,7,8,9,10,11,12,13,14,15);
    const v16h b0 = __builtin_shufflevector(b0lo, b0hi,
        0,1,2,3,4,5,6,7,8,9,10,11,12,13,14,15);
    const v16h b1 = __builtin_shufflevector(b1lo, b1hi,
        0,1,2,3,4,5,6,7,8,9,10,11,12,13,14,15);

    acc00 = __builtin_amdgcn_wmma_f32_16x16x32_f16(
        false, a0, false, b0, (short)0, acc00, false, false);
    acc01 = __builtin_amdgcn_wmma_f32_16x16x32_f16(
        false, a0, false, b1, (short)0, acc01, false, false);
    acc10 = __builtin_amdgcn_wmma_f32_16x16x32_f16(
        false, a1, false, b0, (short)0, acc10, false, false);
    acc11 = __builtin_amdgcn_wmma_f32_16x16x32_f16(
        false, a1, false, b1, (short)0, acc11, false, false);
  }

  // Epilogue: d = sqrt(max(|x|^2 + |w|^2 - 2*dot, 0))
  // C/D layout: VGPR r of lane l -> (M = r + 8*(l>>4), N = l&15)
  auto emit = [&](int mi, int ni, const v8f& acc) {
    const int gn = n0 + nw + 16 * ni + lrow;
    const float wnv = wn[gn];
    #pragma unroll
    for (int r = 0; r < 8; ++r) {
      const int gm = m0 + mw + 16 * mi + halfsel * 8 + r;
      const float d2 = xn[gm] + wnv - 2.0f * acc[r];
      dist[(size_t)gm * NW + gn] = sqrtf(fmaxf(d2, 0.0f));
    }
  };
  emit(0, 0, acc00);
  emit(0, 1, acc01);
  emit(1, 0, acc10);
  emit(1, 1, acc11);
}

// ---------------------------------------------------------------------------
// Kernel 2: BMU argmin per batch row. Distances >= 0, so their float bits
// order correctly as unsigned; pack (bits<<32)|idx so min() keeps the lowest
// index on ties (matches jnp.argmin), independent of scan order.
// One block per batch row; b128 loads.
// ---------------------------------------------------------------------------
__global__ __launch_bounds__(256) void bmu(
    const float* __restrict__ dist, float* __restrict__ bm)
{
  __shared__ unsigned long long red[256];
  const int b = blockIdx.x;
  const int t = threadIdx.x;
  const float4* row4 = (const float4*)(dist + (size_t)b * NW);
  unsigned long long best = ~0ull;
  #pragma unroll 4
  for (int c = 0; c < 16; ++c) {
    const int i4 = c * 256 + t;        // float4 index within the row
    const float4 v = row4[i4];
    const int base = i4 * 4;
    const float e[4] = {v.x, v.y, v.z, v.w};
    #pragma unroll
    for (int j = 0; j < 4; ++j) {
      const unsigned long long p =
          ((unsigned long long)__float_as_uint(e[j]) << 32) |
          (unsigned int)(base + j);
      if (p < best) best = p;
    }
  }
  red[t] = best;
  __syncthreads();
  #pragma unroll
  for (int o = 128; o > 0; o >>= 1) {
    if (t < o && red[t + o] < red[t]) red[t] = red[t + o];
    __syncthreads();
  }
  if (t == 0) {
    const int idx = (int)(red[0] & 0xffffffffu);
    bm[2 * b]     = (float)(idx / COLSN);
    bm[2 * b + 1] = (float)(idx % COLSN);
  }
}

// ---------------------------------------------------------------------------
extern "C" void kernel_launch(void* const* d_in, const int* in_sizes, int n_in,
                              void* d_out, int out_size, void* d_ws, size_t ws_size,
                              hipStream_t stream) {
  const float* x = (const float*)d_in[0];   // [2048, 256] f32
  const float* w = (const float*)d_in[1];   // [128, 128, 256] f32

  // Workspace layout (~9.6 MB): xh | wh | xn | wn
  _Float16* xh = (_Float16*)d_ws;
  _Float16* wh = xh + (size_t)BATCH * FEAT;
  float* xn = (float*)(wh + (size_t)NW * FEAT);
  float* wn = xn + BATCH;

  float* dist = (float*)d_out;                    // [2048 * 16384]
  float* bm   = dist + (size_t)BATCH * NW;        // [2048 * 2]

  const size_t lds_bytes = (size_t)(64 + 128) * LDT * sizeof(_Float16); // 104,448

  convert_norms<<<BATCH + NW, 256, 0, stream>>>(x, w, xh, wh, xn, wn);
  som_dist<<<dim3(BATCH / 64, NW / 128), 256, lds_bytes, stream>>>(
      xh, wh, xn, wn, dist);
  bmu<<<BATCH, 256, 0, stream>>>(dist, bm);
}
